// GNNGuard_35854386987611
// MI455X (gfx1250) — matrix operand order repaired and checked by reading, not compiled, not run on hardware
//
#include <hip/hip_runtime.h>
#include <hip/hip_bf16.h>
#include <math.h>

// ---------------------------------------------------------------------------
// GNNGuard 3-layer GCN for MI455X (gfx1250, wave32).
// Dense h@W via V_WMMA_F32_16X16X32_BF16 (bf16 in, f32 accumulate), with the
// packed weight fragments staged into LDS via async global->LDS copies.
// Edge attention / segment sums via wave32 shuffle reductions + f32 atomics.
// ---------------------------------------------------------------------------

typedef __attribute__((ext_vector_type(16))) __bf16 v16bf;
typedef __attribute__((ext_vector_type(8)))  float  v8f;
typedef __attribute__((ext_vector_type(4)))  int    v4i;

__device__ __forceinline__ float waveRedSum(float v) {
#pragma unroll
  for (int o = 16; o > 0; o >>= 1) v += __shfl_xor(v, o, 32);
  return v;
}
__device__ __forceinline__ float waveRedMax(float v) {
#pragma unroll
  for (int o = 16; o > 0; o >>= 1) v = fmaxf(v, __shfl_xor(v, o, 32));
  return v;
}

// ---- CDNA5 async global->LDS copy (ASYNCcnt path), with safe fallback ----
// Probe-verified signature: (v4i addrspace(1)*, v4i addrspace(3)*, imm, imm)
__device__ __forceinline__ void asyncCopy16(const void* g, void* l) {
#if __has_builtin(__builtin_amdgcn_global_load_async_to_lds_b128)
  __builtin_amdgcn_global_load_async_to_lds_b128(
      (__attribute__((address_space(1))) v4i*)(g),
      (__attribute__((address_space(3))) v4i*)(l), 0, 0);
#else
  *(float4*)l = *(const float4*)g;
#endif
}
__device__ __forceinline__ void asyncWait() {
#if __has_builtin(__builtin_amdgcn_s_wait_asynccnt)
  __builtin_amdgcn_s_wait_asynccnt(0);
#elif __has_builtin(__builtin_amdgcn_global_load_async_to_lds_b128)
  asm volatile("s_wait_asynccnt 0x0" ::: "memory");
#endif
}

// -------------------- generic zero --------------------
__global__ void k_zero(float* __restrict__ p, int n) {
  int i = blockIdx.x * blockDim.x + threadIdx.x;
  if (i < n) p[i] = 0.0f;
}

// -------------------- pack W into per-lane bf16 B fragments -----------------
// B fragment layout (16-bit B, 32x16, wave32): lane l holds column n = l&15;
// element j holds K = (j<8 ? 8h+j : 16+8h+(j-8)), h = l>>4, within the 32-deep K block.
// Flat half index = ((kb*NT + nt)*32 + lane)*16 + j.
__global__ void k_pack_w(const float* __restrict__ W, __bf16* __restrict__ pW,
                         int K, int Nw, int NT, int KB) {
  int idx = blockIdx.x * blockDim.x + threadIdx.x;
  int total = KB * NT * 32 * 16;
  if (idx >= total) return;
  int j    = idx & 15;
  int lane = (idx >> 4) & 31;
  int frag = idx >> 9;
  int nt   = frag % NT;
  int kb   = frag / NT;
  int n = nt * 16 + (lane & 15);
  int k = kb * 32 + ((j & 8) << 1) + ((lane >> 4) << 3) + (j & 7);
  float v = (k < K && n < Nw) ? W[(size_t)k * Nw + n] : 0.0f;
  pW[idx] = (__bf16)v;
}

// -------------------- per-node inverse L2 norm --------------------
__global__ void k_node_rnorm(const float* __restrict__ fea, float* __restrict__ rn,
                             int N, int F) {
  int g = blockIdx.x * blockDim.x + threadIdx.x;
  int i = g >> 5, lane = g & 31;
  if (i >= N) return;
  const float4* p = (const float4*)(fea + (size_t)i * F);
  int n4 = F >> 2;
  float s = 0.0f;
  for (int c = lane; c < n4; c += 32) {
    float4 a = p[c];
    s += a.x * a.x + a.y * a.y + a.z * a.z + a.w * a.w;
  }
  s = waveRedSum(s);
  if (lane == 0) {
    float nr = sqrtf(s);
    rn[i] = (nr == 0.0f) ? 1.0f : 1.0f / nr;  // sklearn: zero rows -> norm 1
  }
}

// -------------------- per-edge thresholded cosine sim --------------------
__global__ void k_edge_w(const float* __restrict__ fea, const int* __restrict__ src,
                         const int* __restrict__ dst, const float* __restrict__ rn,
                         float* __restrict__ w, int E, int F) {
  int g = blockIdx.x * blockDim.x + threadIdx.x;
  int e = g >> 5, lane = g & 31;
  if (e >= E) return;
  int s = src[e], d = dst[e];
  const float4* ps = (const float4*)(fea + (size_t)s * F);
  const float4* pd = (const float4*)(fea + (size_t)d * F);
  int n4 = F >> 2;
  float acc = 0.0f;
  for (int c = lane; c < n4; c += 32) {
    float4 a = ps[c], b = pd[c];
    acc += a.x * b.x + a.y * b.y + a.z * b.z + a.w * b.w;
  }
  acc = waveRedSum(acc);
  if (lane == 0) {
    float sim = acc * rn[s] * rn[d];
    w[e] = (sim >= 0.0f && s != d) ? sim : 0.0f;  // P0 = 0, strip diagonal
  }
}

// -------------------- row L1 sums + nonzero counts (by src) --------------------
__global__ void k_row_accum(const int* __restrict__ src, const float* __restrict__ w,
                            float* __restrict__ rowsum, float* __restrict__ degc, int E) {
  int e = blockIdx.x * blockDim.x + threadIdx.x;
  if (e >= E) return;
  float wv = w[e];
  if (wv != 0.0f) {
    int s = src[e];
    atomicAdd(&rowsum[s], fabsf(wv));
    atomicAdd(&degc[s], 1.0f);
  }
}

// -------------------- self-loop weight sw = exp(1/(deg+1)) --------------------
__global__ void k_node_sw(const float* __restrict__ degc, float* __restrict__ sw, int N) {
  int i = blockIdx.x * blockDim.x + threadIdx.x;
  if (i < N) sw[i] = expf(1.0f / (degc[i] + 1.0f));
}

// -------------------- ew = exp(w/rowsum[src]); accumulate GCN degree ------------
__global__ void k_edge_ew(const int* __restrict__ src, const int* __restrict__ dst,
                          const float* __restrict__ w, const float* __restrict__ rowsum,
                          float* __restrict__ ew, float* __restrict__ gdeg, int E) {
  int e = blockIdx.x * blockDim.x + threadIdx.x;
  if (e >= E) return;
  float wv = w[e];
  float ev = 0.0f;
  if (wv != 0.0f) {
    float rs = rowsum[src[e]];
    if (rs > 0.0f) ev = expf(wv / rs);
  }
  ew[e] = ev;
  if (ev != 0.0f) atomicAdd(&gdeg[dst[e]], ev);
}

// -------------------- dinv = rsqrt(gdeg+sw); self coefficient --------------------
__global__ void k_node_dinv(const float* __restrict__ gdeg, const float* __restrict__ sw,
                            float* __restrict__ dinv, float* __restrict__ selfc, int N) {
  int i = blockIdx.x * blockDim.x + threadIdx.x;
  if (i >= N) return;
  float dv = rsqrtf(gdeg[i] + sw[i]);
  dinv[i] = dv;
  selfc[i] = sw[i] * dv * dv;
}

// -------------------- WMMA GEMM tile body --------------------
// A-fragment layout (16-bit A, 16x32): lane l holds row l&15, K = klo..klo+7
// and klo+16..klo+23 with klo = (l<16 ? 0 : 8) -> four b128 loads per K block,
// bf16-convert in register. B fragments come from LDS (ds_load_b128 pairs).
template <bool FULL>
__device__ __forceinline__ void gemm_tile(const float* __restrict__ A,
                                          const v16bf* __restrict__ lw,
                                          float* __restrict__ C,
                                          int M, int K, int Npad, int KB,
                                          int mt, int ng, int lane) {
  int row = mt * 16 + (lane & 15);
  int kl  = (lane < 16) ? 0 : 8;
  const float* ap = A + (size_t)row * K + kl;

  v8f acc0 = {}, acc1 = {}, acc2 = {}, acc3 = {};
  for (int kb = 0; kb < KB; ++kb) {
    v16bf a;
    if (FULL || row < M) {
      const float4* ap4 = (const float4*)(ap + kb * 32);
      float4 f0 = ap4[0], f1 = ap4[1];   // K = klo .. klo+7
      float4 f2 = ap4[4], f3 = ap4[5];   // K = klo+16 .. klo+23
      if (kb + 1 < KB) __builtin_prefetch(ap + kb * 32 + 32, 0, 3);
      a[0]  = (__bf16)f0.x; a[1]  = (__bf16)f0.y; a[2]  = (__bf16)f0.z; a[3]  = (__bf16)f0.w;
      a[4]  = (__bf16)f1.x; a[5]  = (__bf16)f1.y; a[6]  = (__bf16)f1.z; a[7]  = (__bf16)f1.w;
      a[8]  = (__bf16)f2.x; a[9]  = (__bf16)f2.y; a[10] = (__bf16)f2.z; a[11] = (__bf16)f2.w;
      a[12] = (__bf16)f3.x; a[13] = (__bf16)f3.y; a[14] = (__bf16)f3.z; a[15] = (__bf16)f3.w;
    } else {
#pragma unroll
      for (int j = 0; j < 16; ++j) a[j] = (__bf16)0.0f;
    }
    const v16bf* bp = lw + ((size_t)(kb * 4) * 32 + lane);
    v16bf b0 = bp[0], b1 = bp[32], b2 = bp[64], b3 = bp[96];
    acc0 = __builtin_amdgcn_wmma_f32_16x16x32_bf16(false, a, false, b0, (short)0, acc0, false, false);
    acc1 = __builtin_amdgcn_wmma_f32_16x16x32_bf16(false, a, false, b1, (short)0, acc1, false, false);
    acc2 = __builtin_amdgcn_wmma_f32_16x16x32_bf16(false, a, false, b2, (short)0, acc2, false, false);
    acc3 = __builtin_amdgcn_wmma_f32_16x16x32_bf16(false, a, false, b3, (short)0, acc3, false, false);
  }
  // C/D layout: lanes 0-15 hold M=r, lanes 16-31 hold M=r+8, N = lane&15
  int rowbase = mt * 16 + ((lane >> 4) << 3);
  int nb = ng * 64 + (lane & 15);
#pragma unroll
  for (int r = 0; r < 8; ++r) {
    int rr = rowbase + r;
    if (FULL || rr < M) {
      float* crow = C + (size_t)rr * Npad;
      crow[nb]      = acc0[r];
      crow[nb + 16] = acc1[r];
      crow[nb + 32] = acc2[r];
      crow[nb + 48] = acc3[r];
    }
  }
}

#define GEMM_BLOCK 256

// One block owns one 64-column n-group; its B fragments (KB x 4 tiles, <=64KB)
// are async-staged into LDS once, then 8 waves grid-stride over m-tiles.
__global__ void k_gemm(const float* __restrict__ A, const __bf16* __restrict__ pW,
                       float* __restrict__ C, int M, int K, int Npad,
                       int blocksPerNg) {
  __shared__ __align__(32) char ldsB[64 * 1024];
  int NT = Npad >> 4;
  int NG = Npad >> 6;
  int KB = K >> 5;
  int ng   = blockIdx.x % NG;
  int mblk = blockIdx.x / NG;
  int lane = threadIdx.x & 31;
  int wid  = threadIdx.x >> 5;

  // ---- async-stage this n-group's B fragments (4KB per k-block) ----
  {
    const char* gbase = (const char*)pW;
    int chunks = KB * 256;  // 16B chunks
    for (int c = threadIdx.x; c < chunks; c += GEMM_BLOCK) {
      int kb = c >> 8, wi = c & 255;
      const char* g = gbase + (((size_t)(kb * NT + ng * 4)) << 10) + ((size_t)wi << 4);
      char* l = ldsB + (((size_t)kb) << 12) + ((size_t)wi << 4);
      asyncCopy16(g, l);
    }
    asyncWait();
    __syncthreads();
  }
  const v16bf* lw = (const v16bf*)ldsB;

  int mtiles = (M + 15) >> 4;
  int wavesPerNg = blocksPerNg * (GEMM_BLOCK / 32);
  for (int mt = mblk * (GEMM_BLOCK / 32) + wid; mt < mtiles; mt += wavesPerNg) {
    if (mt * 16 + 16 <= M)
      gemm_tile<true >(A, lw, C, M, K, Npad, KB, mt, ng, lane);
    else
      gemm_tile<false>(A, lw, C, M, K, Npad, KB, mt, ng, lane);
  }
}

// -------------------- aggregation: init with bias + self-loop --------------------
__global__ void k_agg_init(const float* __restrict__ hw, const float* __restrict__ selfc,
                           const float* __restrict__ bias, float* __restrict__ out,
                           int N, int Fo, int ld) {
  int idx = blockIdx.x * blockDim.x + threadIdx.x;
  if (idx >= N * Fo) return;
  int i = idx / Fo, c = idx % Fo;
  out[(size_t)i * ld + c] = bias[c] + selfc[i] * hw[(size_t)i * ld + c];
}

// -------------------- aggregation: edge scatter (D^-1/2 A D^-1/2) --------------
__global__ void k_agg_edge(const int* __restrict__ src, const int* __restrict__ dst,
                           const float* __restrict__ ew, const float* __restrict__ dinv,
                           const float* __restrict__ hw, float* __restrict__ out,
                           int E, int Fo, int ld) {
  int g = blockIdx.x * blockDim.x + threadIdx.x;
  int e = g >> 5, lane = g & 31;
  if (e >= E) return;
  float ev = ew[e];
  if (ev == 0.0f) return;
  int s = src[e], d = dst[e];
  float nw = ev * dinv[s] * dinv[d];
  const float* hs = hw + (size_t)s * ld;
  float* od = out + (size_t)d * ld;
  for (int c = lane; c < Fo; c += 32) atomicAdd(&od[c], nw * hs[c]);
}

// -------------------- ReLU --------------------
__global__ void k_relu(float* __restrict__ p, int n) {
  int i = blockIdx.x * blockDim.x + threadIdx.x;
  if (i < n) p[i] = fmaxf(p[i], 0.0f);
}

// -------------------- log_softmax (wave per node) --------------------
__global__ void k_logsm(const float* __restrict__ h, float* __restrict__ out,
                        int N, int C, int ld) {
  int g = blockIdx.x * blockDim.x + threadIdx.x;
  int i = g >> 5, lane = g & 31;
  if (i >= N) return;
  const float* hr = h + (size_t)i * ld;
  float v0 = (lane      < C) ? hr[lane]      : -INFINITY;
  float v1 = (lane + 32 < C) ? hr[lane + 32] : -INFINITY;
  float m = waveRedMax(fmaxf(v0, v1));
  float s = ((lane < C) ? expf(v0 - m) : 0.0f) + ((lane + 32 < C) ? expf(v1 - m) : 0.0f);
  s = waveRedSum(s);
  float ls = m + logf(s);
  float* orow = out + (size_t)i * C;
  if (lane      < C) orow[lane]      = v0 - ls;
  if (lane + 32 < C) orow[lane + 32] = v1 - ls;
}

// ---------------------------------------------------------------------------
static inline int cdiv_i(int a, int b) { return (a + b - 1) / b; }

extern "C" void kernel_launch(void* const* d_in, const int* in_sizes, int n_in,
                              void* d_out, int out_size, void* d_ws, size_t ws_size,
                              hipStream_t stream) {
  const float* x  = (const float*)d_in[0];
  const int*   ei = (const int*)d_in[1];   // harness contract: integer -> const int*
  const float* W1 = (const float*)d_in[2];
  const float* B1 = (const float*)d_in[3];
  const float* W2 = (const float*)d_in[4];
  const float* B2 = (const float*)d_in[5];
  const float* W3 = (const float*)d_in[6];
  const float* B3 = (const float*)d_in[7];

  int H  = in_sizes[3];
  int D2 = in_sizes[5];
  int Cc = in_sizes[7];
  int F  = in_sizes[2] / H;
  int N  = in_sizes[0] / F;
  int E  = in_sizes[1] / 2;
  const int* src = ei;
  const int* dst = ei + E;

  // -------- workspace carve-out (floats) --------
  float* ws = (float*)d_ws;
  size_t o = 0;
  float* buf0   = ws + o; o += (size_t)N * 256;  // layer1 out / layer3 out
  float* buf1   = ws + o; o += (size_t)N * 256;  // layer2 out
  float* hwb    = ws + o; o += (size_t)N * 256;  // GEMM result (pre-aggregation)
  float* wbuf   = ws + o; o += (size_t)E;        // thresholded cosine sims
  float* ewb    = ws + o; o += (size_t)E;        // exp'd normalized weights
  float* rn     = ws + o; o += (size_t)N;
  float* rowsum = ws + o; o += (size_t)N;        // rowsum/degc/gdeg contiguous ->
  float* degc   = ws + o; o += (size_t)N;        //   single zero pass over 3N
  float* gdeg   = ws + o; o += (size_t)N;
  float* swv    = ws + o; o += (size_t)N;
  float* dinv   = ws + o; o += (size_t)N;
  float* selfc  = ws + o; o += (size_t)N;
  __bf16* pWb   = (__bf16*)(ws + o);             // packed weight fragments (<=256KB)

  const float* Wl[3] = {W1, W2, W3};
  const float* Bl[3] = {B1, B2, B3};
  int Fin[3]  = {F, H, D2};
  int Fout[3] = {H, D2, Cc};
  float* houts[3] = {buf0, buf1, buf0};

  const float* hin = x;
  for (int L = 0; L < 3; ++L) {
    int K  = Fin[L];
    int Fo = Fout[L];
    int Npad = ((Fo + 63) / 64) * 64;
    int NT = Npad / 16, KB = K / 32;

    // pack weights into WMMA B-fragment layout (zero-padded columns)
    int ptot = KB * NT * 32 * 16;
    k_pack_w<<<cdiv_i(ptot, 256), 256, 0, stream>>>(Wl[L], pWb, K, Fo, NT, KB);

    // attention coefficients
    k_node_rnorm<<<cdiv_i(N * 32, 256), 256, 0, stream>>>(hin, rn, N, K);
    k_edge_w<<<cdiv_i(E * 32, 256), 256, 0, stream>>>(hin, src, dst, rn, wbuf, E, K);
    k_zero<<<cdiv_i(3 * N, 256), 256, 0, stream>>>(rowsum, 3 * N);
    k_row_accum<<<cdiv_i(E, 256), 256, 0, stream>>>(src, wbuf, rowsum, degc, E);
    k_node_sw<<<cdiv_i(N, 256), 256, 0, stream>>>(degc, swv, N);
    k_edge_ew<<<cdiv_i(E, 256), 256, 0, stream>>>(src, dst, wbuf, rowsum, ewb, gdeg, E);
    k_node_dinv<<<cdiv_i(N, 256), 256, 0, stream>>>(gdeg, swv, dinv, selfc, N);

    // dense h @ W via bf16 WMMA (B staged in LDS, async copies)
    int NG = Npad / 64;
    int blocksPerNg = 128;
    k_gemm<<<NG * blocksPerNg, GEMM_BLOCK, 0, stream>>>(hin, pWb, hwb, N, K, Npad,
                                                        blocksPerNg);

    // GCN aggregation
    float* hout = houts[L];
    k_agg_init<<<cdiv_i(N * Fo, 256), 256, 0, stream>>>(hwb, selfc, Bl[L], hout, N, Fo, Npad);
    k_agg_edge<<<cdiv_i(E * 32, 256), 256, 0, stream>>>(src, dst, ewb, dinv, hwb, hout, E, Fo, Npad);

    if (L < 2) {
      k_relu<<<cdiv_i(N * Fo, 256), 256, 0, stream>>>(hout, N * Fo);
      hin = hout;
    } else {
      k_logsm<<<cdiv_i(N * 32, 256), 256, 0, stream>>>(hout, (float*)d_out, N, Fo, Npad);
    }
  }
}
